// PVDST_partseg_11673721111028
// MI455X (gfx1250) — matrix operand order repaired and checked by reading, not compiled, not run on hardware
//
#include <hip/hip_runtime.h>
#include <hip/hip_bf16.h>
#include <math.h>

// ---------------------------------------------------------------------------
// Types for CDNA5 WMMA (wave32, 16x16x32 bf16 -> f32)
// ---------------------------------------------------------------------------
typedef __attribute__((ext_vector_type(16))) __bf16 v16bf;
typedef __attribute__((ext_vector_type(8)))  __bf16 v8bf;
typedef __attribute__((ext_vector_type(8)))  float  v8f;

static constexpr int BK  = 16;        // batch
static constexpr int NK  = 2048;      // points per batch
static constexpr int PK  = BK * NK;   // 32768 total point-columns
static constexpr int KNN = 16;

// ---------------------------------------------------------------------------
// 1) Pack input points: (B,6,N) f32 -> act0[p][32] bf16 (zero padded) and
//    xyz4[p] = {x,y,z,|p|^2} f32 for the KNN.
// ---------------------------------------------------------------------------
__global__ __launch_bounds__(256) void pack_points_kernel(
    const float* __restrict__ pts, __bf16* __restrict__ act0,
    float4* __restrict__ xyz4) {
  int p = blockIdx.x * 256 + threadIdx.x;
  if (p >= PK) return;
  int b = p >> 11, n = p & 2047;
  const float* src = pts + (size_t)b * 6 * NK + n;
  float c[6];
#pragma unroll
  for (int j = 0; j < 6; ++j) c[j] = src[(size_t)j * NK];
  __bf16* d = act0 + (size_t)p * 32;
#pragma unroll
  for (int j = 0; j < 6; ++j) d[j] = (__bf16)c[j];
#pragma unroll
  for (int j = 6; j < 32; ++j) d[j] = (__bf16)0.0f;
  xyz4[p] = make_float4(c[0], c[1], c[2],
                        c[0] * c[0] + c[1] * c[1] + c[2] * c[2]);
}

// ---------------------------------------------------------------------------
// 2) KNN: per-batch 16 nearest neighbors (self included, order irrelevant —
//    only consumed by gather+max). One thread per query; 256-point LDS tiles;
//    register-resident top-16 via replace-max (unrolled -> cndmask selects).
//    d(n,m) ranking uses |pm|^2 - 2 pn.pm (query norm is constant).
// ---------------------------------------------------------------------------
__global__ __launch_bounds__(256) void knn_kernel(
    const float4* __restrict__ xyz4, int* __restrict__ idx) {
  __shared__ float4 tile[256];
  const int b = blockIdx.y;
  const int n = blockIdx.x * 256 + threadIdx.x;
  const float4 q = xyz4[(size_t)b * NK + n];
  float bd[KNN];
  int   bi[KNN];
#pragma unroll
  for (int j = 0; j < KNN; ++j) { bd[j] = 3.0e38f; bi[j] = 0; }
  float worst = 3.0e38f; int wslot = 0;
  for (int t = 0; t < NK; t += 256) {
    __syncthreads();
    tile[threadIdx.x] = xyz4[(size_t)b * NK + t + threadIdx.x];
    __syncthreads();
    for (int m = 0; m < 256; ++m) {
      float4 c = tile[m];
      float d = c.w - 2.0f * (q.x * c.x + q.y * c.y + q.z * c.z);
      if (d < worst) {
#pragma unroll
        for (int j = 0; j < KNN; ++j)
          if (j == wslot) { bd[j] = d; bi[j] = t + m; }
        worst = bd[0]; wslot = 0;
#pragma unroll
        for (int j = 1; j < KNN; ++j)
          if (bd[j] > worst) { worst = bd[j]; wslot = j; }
      }
    }
  }
  int* op = idx + ((size_t)b * NK + n) * KNN;
#pragma unroll
  for (int j = 0; j < KNN; ++j) op[j] = bi[j];
}

// ---------------------------------------------------------------------------
// 3) Weight packing into a WMMA A-fragment layout (contiguous k, 16-bit A
//    16x32):  lane l, half j: M = l&15 ; v=j>>1, h=j&1 ;
//    K = j + (l>=16 ? 16 : 0).
//    Rows are pre-scaled by the BN gamma (y = (gW)x + beta).
//    dst index = ((o_tile*nK + k_tile)*32 + lane)*16 + j.
// ---------------------------------------------------------------------------
__global__ __launch_bounds__(256) void pack_wmma_a_kernel(
    __bf16* __restrict__ dst, const float* __restrict__ W,
    const float* __restrict__ g, int O, int Ovalid, int srcStride,
    int Kvalid, int Kp) {
  int id = blockIdx.x * 256 + threadIdx.x;
  int total = O * Kp;
  if (id >= total) return;
  int j = id & 15;
  int lane = (id >> 4) & 31;
  int tl = id >> 9;          // o_tile * nK + k_tile
  int nKt = Kp >> 5;
  int kt = tl % nKt, ot = tl / nKt;
  int M = lane & 15;
  int v = j >> 1, hb = j & 1;
  int inner = j;
  int Kk = inner + ((lane >= 16) ? 16 : 0);
  int orow = ot * 16 + M;
  int kcol = kt * 32 + Kk;
  float val = 0.0f;
  if (orow < Ovalid && kcol < Kvalid) {
    val = W[(size_t)orow * srcStride + kcol];
    if (g) val *= g[orow];
  }
  dst[id] = (__bf16)val;
}

// Fused bias: dst[o] = g[o]*convb[o] + beta[o] with null handling, 0 beyond
// Ovalid (used for conv-bias+BN layers and the padded final layer).
__global__ __launch_bounds__(256) void fuse_bias_kernel(
    float* __restrict__ dst, const float* __restrict__ g,
    const float* __restrict__ convb, const float* __restrict__ beta,
    int O, int Ovalid) {
  int o = blockIdx.x * 256 + threadIdx.x;
  if (o >= O) return;
  float v = 0.0f;
  if (o < Ovalid) {
    float cb = convb ? convb[o] : 0.0f;
    float gg = g ? g[o] : 1.0f;
    v = gg * cb + (beta ? beta[o] : 0.0f);
  }
  dst[o] = v;
}

// ---------------------------------------------------------------------------
// 4) Generic WMMA GEMM: Y[p][o] = epi( (gW) @ X[p][:] + bias (+res) ).
//    X is point-major bf16 with (stride, offset); B-fragment, contiguous k:
//    lane l -> column l&15 of the 16-point tile, K offset (l<16?0:16),
//    16 contiguous halves. One wave owns a 16-point x 64-channel tile
//    (4 M-tiles reuse one B fragment per K step).
// ---------------------------------------------------------------------------
enum { EPI_RELU = 0, EPI_LEAKY = 1, EPI_SEG_F32 = 2 };

template <int EPI, bool RESIDUAL, bool PER_BATCH_BIAS>
__global__ __launch_bounds__(256) void wmma_gemm_kernel(
    const __bf16* __restrict__ Apack,
    const __bf16* __restrict__ X, int xs, int xo,
    __bf16* __restrict__ Y, int ys, int yo,
    const float* __restrict__ bias,
    const __bf16* __restrict__ Res, int rs, int ro,
    float* __restrict__ outF32,
    int O, int nK) {
  const int lane = threadIdx.x & 31;
  const int unit = blockIdx.x * 8 + (threadIdx.x >> 5);
  const int nPt = PK / 16;                 // 2048 point tiles
  const int pt = unit & (nPt - 1);
  const int og = unit / nPt;               // 64-channel group
  const int col = lane & 15;
  const int khalf = (lane < 16) ? 0 : 16;

  const __bf16* xcol = X + (size_t)(pt * 16 + col) * xs + xo + khalf;
  const __bf16* abase = Apack + (size_t)og * 4 * nK * 512 + lane * 16;

  v8f acc0 = {}, acc1 = {}, acc2 = {}, acc3 = {};
  for (int k = 0; k < nK; ++k) {
    v16bf bfrag = *(const v16bf*)(xcol + (size_t)k * 32);
    const __bf16* ak = abase + (size_t)k * 512;
    v16bf a0 = *(const v16bf*)(ak);
    v16bf a1 = *(const v16bf*)(ak + (size_t)nK * 512);
    v16bf a2 = *(const v16bf*)(ak + (size_t)2 * nK * 512);
    v16bf a3 = *(const v16bf*)(ak + (size_t)3 * nK * 512);
    if (k + 1 < nK) {                       // gfx1250 global_prefetch_b8
      __builtin_prefetch(xcol + (size_t)(k + 1) * 32, 0, 1);
      __builtin_prefetch(ak + 512, 0, 1);
    }
    acc0 = __builtin_amdgcn_wmma_f32_16x16x32_bf16(false, a0, false, bfrag,
                                                   (short)0, acc0, false, false);
    acc1 = __builtin_amdgcn_wmma_f32_16x16x32_bf16(false, a1, false, bfrag,
                                                   (short)0, acc1, false, false);
    acc2 = __builtin_amdgcn_wmma_f32_16x16x32_bf16(false, a2, false, bfrag,
                                                   (short)0, acc2, false, false);
    acc3 = __builtin_amdgcn_wmma_f32_16x16x32_bf16(false, a3, false, bfrag,
                                                   (short)0, acc3, false, false);
  }

  // C/D layout: lane l, vgpr r -> (M = r + (l<16?0:8), N = l&15)
  const int prow = pt * 16 + col;
  const int orow0 = og * 64 + ((lane < 16) ? 0 : 8);
  const float* bp = bias + (PER_BATCH_BIAS ? (size_t)(prow >> 11) * O : 0);
  v8f accs[4] = {acc0, acc1, acc2, acc3};

#pragma unroll
  for (int mt = 0; mt < 4; ++mt) {
    const int ob = orow0 + mt * 16;
    v8f vacc = accs[mt];
    float resv[8];
    if (RESIDUAL) {
      v8bf rv = *(const v8bf*)(Res + (size_t)prow * rs + ro + ob);
#pragma unroll
      for (int r = 0; r < 8; ++r) resv[r] = (float)rv[r];
    }
    if (EPI == EPI_SEG_F32) {
      int b = prow >> 11, n = prow & 2047;
#pragma unroll
      for (int r = 0; r < 8; ++r) {
        int o = ob + r;
        if (o < 50) {
          float v = vacc[r] + bp[o];
          outF32[((size_t)b * 50 + o) * 2048 + n] = v;
        }
      }
    } else {
      v8bf outv;
#pragma unroll
      for (int r = 0; r < 8; ++r) {
        float v = vacc[r] + bp[ob + r];
        if (RESIDUAL) v += resv[r];
        if (EPI == EPI_RELU) v = fmaxf(v, 0.0f);
        else                 v = (v > 0.0f) ? v : 0.2f * v;
        outv[r] = (__bf16)v;
      }
      *(v8bf*)(Y + (size_t)prow * ys + yo + ob) = outv;
    }
  }
}

// ---------------------------------------------------------------------------
// 5) Gather + max over 16 neighbors. One thread = one point x 8 channels.
// ---------------------------------------------------------------------------
__global__ __launch_bounds__(256) void gather_max_kernel(
    const __bf16* __restrict__ t, const int* __restrict__ idx,
    __bf16* __restrict__ m) {
  int id = blockIdx.x * 256 + threadIdx.x;  // PK*16 work items
  int cg = id & 15, p = id >> 4;
  if (p >= PK) return;
  int base = (p >> 11) << 11;               // b*NK
  const int* ip = idx + (size_t)p * KNN;
  float acc[8];
#pragma unroll
  for (int r = 0; r < 8; ++r) acc[r] = -3.0e38f;
  for (int k = 0; k < KNN; ++k) {
    int nb = ip[k];
    v8bf vv = *(const v8bf*)(t + (size_t)(base + nb) * 128 + cg * 8);
#pragma unroll
    for (int r = 0; r < 8; ++r) acc[r] = fmaxf(acc[r], (float)vv[r]);
  }
  v8bf o;
#pragma unroll
  for (int r = 0; r < 8; ++r) o[r] = (__bf16)acc[r];
  *(v8bf*)(m + (size_t)p * 128 + cg * 8) = o;
}

// ---------------------------------------------------------------------------
// 6) Global max / mean over N for xf (stride 1024). Coalesced across o.
// ---------------------------------------------------------------------------
__global__ __launch_bounds__(256) void reduce_xf_kernel(
    const __bf16* __restrict__ xf, float* __restrict__ gmax,
    float* __restrict__ gavg) {
  int b = blockIdx.x >> 2;
  int o = ((blockIdx.x & 3) << 8) + threadIdx.x;
  const __bf16* base = xf + (size_t)b * NK * 1024 + o;
  float mx = -3.0e38f, sm = 0.0f;
  for (int n = 0; n < NK; ++n) {
    float v = (float)base[(size_t)n * 1024];
    mx = fmaxf(mx, v);
    sm += v;
  }
  gmax[b * 1024 + o] = mx;
  gavg[b * 1024 + o] = sm * (1.0f / 2048.0f);
}

// 7) Label branch: lab[b][o] = leaky(gl*(Wl@label[b]) + bl), constant over N.
__global__ __launch_bounds__(256) void labf_kernel(
    const float* __restrict__ Wl, const float* __restrict__ gl,
    const float* __restrict__ bl, const float* __restrict__ label,
    float* __restrict__ labf) {
  int id = blockIdx.x * 256 + threadIdx.x;
  if (id >= BK * 64) return;
  int b = id >> 6, o = id & 63;
  float s = 0.0f;
#pragma unroll
  for (int c = 0; c < 16; ++c) s += Wl[o * 16 + c] * label[b * 16 + c];
  s = s * gl[o] + bl[o];
  labf[id] = (s > 0.0f) ? s : 0.2f * s;
}

// 8) Fold the per-batch-constant columns of Wc1 (x_max/x_avg/lab broadcasts)
//    plus conv bias + BN into a per-batch bias: shrinks the big GEMM K
//    from 3136 to 1024.
__global__ __launch_bounds__(256) void cbias_kernel(
    const float* __restrict__ Wc1, const float* __restrict__ gc1,
    const float* __restrict__ bc1, const float* __restrict__ bb1,
    const float* __restrict__ gmax, const float* __restrict__ gavg,
    const float* __restrict__ labf, float* __restrict__ cbias) {
  int id = blockIdx.x * 256 + threadIdx.x;
  if (id >= BK * 512) return;
  int b = id >> 9, o = id & 511;
  const float* wr = Wc1 + (size_t)o * 3136;
  const float* gm = gmax + (size_t)b * 1024;
  const float* ga = gavg + (size_t)b * 1024;
  float s = 0.0f;
  for (int c = 0; c < 1024; ++c)
    s += wr[1024 + c] * gm[c] + wr[2048 + c] * ga[c];
  for (int j = 0; j < 64; ++j) s += wr[3072 + j] * labf[b * 64 + j];
  cbias[id] = gc1[o] * (s + bc1[o]) + bb1[o];
}

// ---------------------------------------------------------------------------
// Host driver
// ---------------------------------------------------------------------------
extern "C" void kernel_launch(void* const* d_in, const int* in_sizes, int n_in,
                              void* d_out, int out_size, void* d_ws,
                              size_t ws_size, hipStream_t stream) {
  (void)in_sizes; (void)n_in; (void)out_size; (void)ws_size;

  const float* points = (const float*)d_in[0];
  const float* label  = (const float*)d_in[1];
  const float* We1 = (const float*)d_in[2];
  const float* ge1 = (const float*)d_in[3];
  const float* be1 = (const float*)d_in[4];
  const float* We2 = (const float*)d_in[5];
  const float* ge2 = (const float*)d_in[6];
  const float* be2 = (const float*)d_in[7];
  const float* W1  = (const float*)d_in[8];
  const float* g1  = (const float*)d_in[9];
  const float* b1  = (const float*)d_in[10];
  const float* W2  = (const float*)d_in[11];
  const float* g2  = (const float*)d_in[12];
  const float* b2  = (const float*)d_in[13];
  const float* Wf  = (const float*)d_in[14];
  const float* gf  = (const float*)d_in[15];
  const float* bfv = (const float*)d_in[16];
  const float* Wl  = (const float*)d_in[17];
  const float* gl  = (const float*)d_in[18];
  const float* bl  = (const float*)d_in[19];
  const float* Wc1 = (const float*)d_in[20];
  const float* bc1 = (const float*)d_in[21];
  const float* gc1 = (const float*)d_in[22];
  const float* bb1 = (const float*)d_in[23];
  const float* Wc2 = (const float*)d_in[24];
  const float* bc2 = (const float*)d_in[25];
  const float* gc2 = (const float*)d_in[26];
  const float* bb2 = (const float*)d_in[27];
  const float* Wc3 = (const float*)d_in[28];
  const float* bc3 = (const float*)d_in[29];
  float* outp = (float*)d_out;

  // --- workspace carve-out ---
  char* base = (char*)d_ws;
  size_t off = 0;
  auto alloc = [&](size_t bytes) -> char* {
    char* p = base + off;
    off = (off + bytes + 255) & ~(size_t)255;
    return p;
  };
  __bf16* act0 = (__bf16*)alloc((size_t)PK * 32 * 2);
  __bf16* bufA = (__bf16*)alloc((size_t)PK * 128 * 2);
  __bf16* bufB = (__bf16*)alloc((size_t)PK * 128 * 2);
  __bf16* mbuf = (__bf16*)alloc((size_t)PK * 128 * 2);
  __bf16* xcat = (__bf16*)alloc((size_t)PK * 384 * 2);
  __bf16* xfb  = (__bf16*)alloc((size_t)PK * 1024 * 2);
  __bf16* c1b  = (__bf16*)alloc((size_t)PK * 512 * 2);
  __bf16* c2b  = (__bf16*)alloc((size_t)PK * 256 * 2);
  float4* xyz4 = (float4*)alloc((size_t)PK * 16);
  int*    idxb = (int*)alloc((size_t)PK * KNN * 4);
  float*  gmax = (float*)alloc((size_t)BK * 1024 * 4);
  float*  gavg = (float*)alloc((size_t)BK * 1024 * 4);
  float*  labf = (float*)alloc((size_t)BK * 64 * 4);
  float*  cbia = (float*)alloc((size_t)BK * 512 * 4);
  float*  bWc2 = (float*)alloc(256 * 4);
  float*  bWc3 = (float*)alloc(64 * 4);
  __bf16* pWe1 = (__bf16*)alloc((size_t)128 * 32 * 2);
  __bf16* pWe2 = (__bf16*)alloc((size_t)128 * 128 * 2);
  __bf16* pW1  = (__bf16*)alloc((size_t)3 * 128 * 128 * 2);
  __bf16* pW2  = (__bf16*)alloc((size_t)3 * 128 * 128 * 2);
  __bf16* pWf  = (__bf16*)alloc((size_t)1024 * 384 * 2);
  __bf16* pWc1 = (__bf16*)alloc((size_t)512 * 1024 * 2);
  __bf16* pWc2 = (__bf16*)alloc((size_t)256 * 512 * 2);
  __bf16* pWc3 = (__bf16*)alloc((size_t)64 * 256 * 2);

  auto blk = [](int total) { return (total + 255) / 256; };
  auto ggrid = [](int O) { return (PK / 16) * (O / 64) / 8; };

  // --- stage inputs ---
  pack_points_kernel<<<blk(PK), 256, 0, stream>>>(points, act0, xyz4);
  knn_kernel<<<dim3(NK / 256, BK), 256, 0, stream>>>(xyz4, idxb);

  // --- pack all weights (BN gamma folded into rows) ---
  pack_wmma_a_kernel<<<blk(128 * 32), 256, 0, stream>>>(pWe1, We1, ge1, 128, 128, 6, 6, 32);
  pack_wmma_a_kernel<<<blk(128 * 128), 256, 0, stream>>>(pWe2, We2, ge2, 128, 128, 128, 128, 128);
  for (int i = 0; i < 3; ++i) {
    pack_wmma_a_kernel<<<blk(128 * 128), 256, 0, stream>>>(
        pW1 + (size_t)i * 128 * 128, W1 + (size_t)i * 128 * 128, g1 + i * 128,
        128, 128, 128, 128, 128);
    pack_wmma_a_kernel<<<blk(128 * 128), 256, 0, stream>>>(
        pW2 + (size_t)i * 128 * 128, W2 + (size_t)i * 128 * 128, g2 + i * 128,
        128, 128, 128, 128, 128);
  }
  pack_wmma_a_kernel<<<blk(1024 * 384), 256, 0, stream>>>(pWf, Wf, gf, 1024, 1024, 384, 384, 384);
  pack_wmma_a_kernel<<<blk(512 * 1024), 256, 0, stream>>>(pWc1, Wc1, gc1, 512, 512, 3136, 1024, 1024);
  pack_wmma_a_kernel<<<blk(256 * 512), 256, 0, stream>>>(pWc2, Wc2, gc2, 256, 256, 512, 512, 512);
  pack_wmma_a_kernel<<<blk(64 * 256), 256, 0, stream>>>(pWc3, Wc3, nullptr, 64, 50, 256, 256, 256);
  fuse_bias_kernel<<<1, 256, 0, stream>>>(bWc2, gc2, bc2, bb2, 256, 256);
  fuse_bias_kernel<<<1, 256, 0, stream>>>(bWc3, nullptr, bc3, nullptr, 64, 50);

  // --- embedding: h = relu(bn(We2 @ relu(bn(We1 @ x)))) ---
  wmma_gemm_kernel<EPI_RELU, false, false><<<ggrid(128), 256, 0, stream>>>(
      pWe1, act0, 32, 0, bufA, 128, 0, be1, nullptr, 0, 0, nullptr, 128, 1);
  wmma_gemm_kernel<EPI_RELU, false, false><<<ggrid(128), 256, 0, stream>>>(
      pWe2, bufA, 128, 0, bufB, 128, 0, be2, nullptr, 0, 0, nullptr, 128, 4);

  // --- 3 EdgeConv rounds; h_i lives in xcat slice i ---
  for (int i = 0; i < 3; ++i) {
    const __bf16* hin = (i == 0) ? bufB : xcat;
    int hs = (i == 0) ? 128 : 384;
    int ho = (i == 0) ? 0 : 128 * (i - 1);
    // t = relu(bn(W1[i] @ h))
    wmma_gemm_kernel<EPI_RELU, false, false><<<ggrid(128), 256, 0, stream>>>(
        pW1 + (size_t)i * 128 * 128, hin, hs, ho, bufA, 128, 0, b1 + i * 128,
        nullptr, 0, 0, nullptr, 128, 4);
    // m = max over 16 neighbors
    gather_max_kernel<<<blk(PK * 16), 256, 0, stream>>>(bufA, idxb, mbuf);
    // h = relu(bn(W2[i] @ m) + h)  -> xcat slice i
    wmma_gemm_kernel<EPI_RELU, true, false><<<ggrid(128), 256, 0, stream>>>(
        pW2 + (size_t)i * 128 * 128, mbuf, 128, 0, xcat, 384, 128 * i,
        b2 + i * 128, hin, hs, ho, nullptr, 128, 4);
  }

  // --- xf = leaky_relu(bn(Wf @ xcat), 0.2) ---
  wmma_gemm_kernel<EPI_LEAKY, false, false><<<ggrid(1024), 256, 0, stream>>>(
      pWf, xcat, 384, 0, xfb, 1024, 0, bfv, nullptr, 0, 0, nullptr, 1024, 12);

  // --- pooled stats, label branch, per-batch bias fold for Wc1 ---
  reduce_xf_kernel<<<BK * 4, 256, 0, stream>>>(xfb, gmax, gavg);
  labf_kernel<<<blk(BK * 64), 256, 0, stream>>>(Wl, gl, bl, label, labf);
  cbias_kernel<<<blk(BK * 512), 256, 0, stream>>>(Wc1, gc1, bc1, bb1, gmax,
                                                  gavg, labf, cbia);

  // --- head MLP ---
  wmma_gemm_kernel<EPI_RELU, false, true><<<ggrid(512), 256, 0, stream>>>(
      pWc1, xfb, 1024, 0, c1b, 512, 0, cbia, nullptr, 0, 0, nullptr, 512, 32);
  wmma_gemm_kernel<EPI_RELU, false, false><<<ggrid(256), 256, 0, stream>>>(
      pWc2, c1b, 512, 0, c2b, 256, 0, bWc2, nullptr, 0, 0, nullptr, 256, 16);
  wmma_gemm_kernel<EPI_SEG_F32, false, false><<<ggrid(64), 256, 0, stream>>>(
      pWc3, c2b, 256, 0, nullptr, 0, 0, bWc3, nullptr, 0, 0, outp, 64, 8);
}
